// SAAttnMem_48438641164633
// MI455X (gfx1250) — matrix operand order repaired and checked by reading, not compiled, not run on hardware
//
#include <hip/hip_runtime.h>
#include <hip/hip_bf16.h>
#include <math.h>

// Problem constants (from reference): N=4, CIN=64, DM=32, H=W=64, S=4096
#define BN   4
#define CIN  64
#define DM   32
#define HW   64
#define SS   4096          // H*W
#define NS   (BN*SS)       // 16384 spatial positions total

typedef __attribute__((ext_vector_type(16))) _Float16 v16h;
typedef __attribute__((ext_vector_type(8)))  _Float16 v8h;
typedef __attribute__((ext_vector_type(8)))  float    v8f;

union V16U { v16h v; v8h h[2]; };

// DPP16 xor-style all-reduce (max) across each 16-lane row of the wave.
// Sequence: quad_perm(1,0,3,2), quad_perm(2,3,0,1), row_half_mirror, row_mirror.
// After steps 1-2 values are uniform in groups of 4, so the mirrors act as
// xor4 / xor8 at group granularity -> all 16 lanes hold the row max.
template <int CTRL>
__device__ __forceinline__ float fmax_dpp(float x) {
    int s = __builtin_bit_cast(int, x);
    int y = __builtin_amdgcn_update_dpp(s, s, CTRL, 0xF, 0xF, true);
    return fmaxf(x, __builtin_bit_cast(float, y));
}
__device__ __forceinline__ float rowmax16(float x) {
    x = fmax_dpp<0xB1>(x);   // quad_perm [1,0,3,2]  (xor 1)
    x = fmax_dpp<0x4E>(x);   // quad_perm [2,3,0,1]  (xor 2)
    x = fmax_dpp<0x141>(x);  // row_half_mirror       (xor 4 at group level)
    x = fmax_dpp<0x140>(x);  // row_mirror            (xor 8 at group level)
    return x;
}

// ---------------------------------------------------------------------------
// Kernel 1: 1x1 projections (f32), emit q/k/v in f16 in WMMA-friendly layouts
//   q,kh,km : [n][s][32] f16   (c contiguous)   q pre-scaled by 1/sqrt(32)
//   vh,vm   : [n][c][s]  f16   (s contiguous)
// ---------------------------------------------------------------------------
__global__ __launch_bounds__(128) void proj_kernel(
    const float* __restrict__ h, const float* __restrict__ m,
    const float* __restrict__ Wh, const float* __restrict__ bh,
    const float* __restrict__ Wm, const float* __restrict__ bm,
    _Float16* __restrict__ q,  _Float16* __restrict__ kh,
    _Float16* __restrict__ vh, _Float16* __restrict__ km,
    _Float16* __restrict__ vm)
{
    __shared__ float xs[CIN * 128];           // 32 KB input tile
    const int tid = threadIdx.x;
    const int idx = blockIdx.x * 128 + tid;   // global (n,s)
    const int n   = idx >> 12;
    const int sp  = idx & (SS - 1);
    const float qscale = 0.17677669529663687f;   // 1/sqrt(32)

    // ---- h projections (96 outputs) ----
    for (int ci = 0; ci < CIN; ++ci)
        xs[ci * 128 + tid] = h[((size_t)(n * CIN + ci)) * SS + sp];
    __syncthreads();

    for (int co = 0; co < 3 * DM; ++co) {
        float acc = bh[co];
        const float* wr = Wh + co * CIN;
        #pragma unroll 8
        for (int ci = 0; ci < CIN; ++ci)
            acc += wr[ci] * xs[ci * 128 + tid];
        if (co < DM)            q [(size_t)idx * DM + co]            = (_Float16)(acc * qscale);
        else if (co < 2 * DM)   kh[(size_t)idx * DM + (co - DM)]     = (_Float16)acc;
        else                    vh[((size_t)(n * DM + (co - 2*DM))) * SS + sp] = (_Float16)acc;
    }
    __syncthreads();

    // ---- m projections (64 outputs) ----
    for (int ci = 0; ci < CIN; ++ci)
        xs[ci * 128 + tid] = m[((size_t)(n * CIN + ci)) * SS + sp];
    __syncthreads();

    for (int co = 0; co < 2 * DM; ++co) {
        float acc = bm[co];
        const float* wr = Wm + co * CIN;
        #pragma unroll 8
        for (int ci = 0; ci < CIN; ++ci)
            acc += wr[ci] * xs[ci * 128 + tid];
        if (co < DM)  km[(size_t)idx * DM + co]                      = (_Float16)acc;
        else          vm[((size_t)(n * DM + (co - DM))) * SS + sp]   = (_Float16)acc;
    }
}

// ---------------------------------------------------------------------------
// Kernel 2: flash attention, one wave per 16-row query block, 64 columns per
// iteration.  Score tiles via v_wmma_f32_16x16x32_f16; online softmax with
// DPP16 row-max; row sums via an extra WMMA against an all-ones B matrix
// (replaces the cross-lane sum butterfly); P transposed through LDS into the
// f16 A-operand layout for the PV WMMAs.
// z out: [n][c][s] f32
// ---------------------------------------------------------------------------
__global__ __launch_bounds__(32) void attn_kernel(
    const _Float16* __restrict__ qm, const _Float16* __restrict__ kmat,
    const _Float16* __restrict__ vmat, float* __restrict__ z)
{
    __shared__ __align__(32) _Float16 ph[16 * 72];   // 16x64 P tile, padded rows

    const int lane = threadIdx.x;
    const int lm   = lane & 15;
    const int hi   = lane >> 4;              // 0: lanes 0-15, 1: lanes 16-31
    const int n    = blockIdx.x >> 8;        // 256 q-blocks per batch
    const int qb   = blockIdx.x & 255;
    const int s0   = qb << 4;

    // A operand (q block): lane = M row; contiguous K order, k = 16*hi + s.
    const _Float16* qr = qm + ((size_t)(n * SS + s0 + lm)) * DM;
    V16U a;
    a.h[0] = *(const v8h*)(qr + (hi ? 16 : 0));    // K 0-7  | 16-23
    a.h[1] = *(const v8h*)(qr + (hi ? 24 : 8));    // K 8-15 | 24-31

    const _Float16* kb = kmat + (size_t)n * SS * DM;
    const _Float16* vb = vmat + (size_t)n * DM * SS;

    // All-ones B operand for the row-sum WMMA.
    v16h ones;
    #pragma unroll
    for (int i = 0; i < 16; ++i) ones[i] = (_Float16)1.0f;

    v8f acc0 = {0.f,0.f,0.f,0.f,0.f,0.f,0.f,0.f};   // c = 0..15
    v8f acc1 = {0.f,0.f,0.f,0.f,0.f,0.f,0.f,0.f};   // c = 16..31
    v8f accL = {0.f,0.f,0.f,0.f,0.f,0.f,0.f,0.f};   // row sums of P
    float mr[8];
    #pragma unroll
    for (int r = 0; r < 8; ++r) mr[r] = -3.0e38f;

    for (int t0 = 0; t0 < SS; t0 += 64) {
        // ---- score tiles: columns t0 .. t0+63 ----
        v16h b0 = *(const v16h*)(kb + ((size_t)(t0      + lm)) * DM + (hi ? 16 : 0));
        v16h b1 = *(const v16h*)(kb + ((size_t)(t0 + 16 + lm)) * DM + (hi ? 16 : 0));
        v16h b2 = *(const v16h*)(kb + ((size_t)(t0 + 32 + lm)) * DM + (hi ? 16 : 0));
        v16h b3 = *(const v16h*)(kb + ((size_t)(t0 + 48 + lm)) * DM + (hi ? 16 : 0));
        if (t0 + 64 < SS) {
            __builtin_prefetch(kb + ((size_t)(t0 + 64 + lm)) * DM, 0, 1);
            __builtin_prefetch(vb + (size_t)lm * SS + t0 + 64, 0, 1);
        }
        v8f zc = {0.f,0.f,0.f,0.f,0.f,0.f,0.f,0.f};
        v8f sA = __builtin_amdgcn_wmma_f32_16x16x32_f16(false, a.v, false, b0, (short)0, zc, false, false);
        v8f sB = __builtin_amdgcn_wmma_f32_16x16x32_f16(false, a.v, false, b1, (short)0, zc, false, false);
        v8f sC = __builtin_amdgcn_wmma_f32_16x16x32_f16(false, a.v, false, b2, (short)0, zc, false, false);
        v8f sD = __builtin_amdgcn_wmma_f32_16x16x32_f16(false, a.v, false, b3, (short)0, zc, false, false);

        // ---- online softmax: element r = row (r + 8*hi), column = lane ----
        #pragma unroll
        for (int r = 0; r < 8; ++r) {
            float tmax = fmaxf(fmaxf(sA[r], sB[r]), fmaxf(sC[r], sD[r]));
            tmax = rowmax16(tmax);
            const float nm = fmaxf(mr[r], tmax);
            const float sc = __expf(mr[r] - nm);
            mr[r] = nm;
            acc0[r] *= sc;
            acc1[r] *= sc;
            accL[r] *= sc;
            sA[r] = __expf(sA[r] - nm);
            sB[r] = __expf(sB[r] - nm);
            sC[r] = __expf(sC[r] - nm);
            sD[r] = __expf(sD[r] - nm);
        }

        // ---- transpose P (16x64) through LDS into f16 A-operand layout ----
        __syncthreads();
        {
            const int rowb = hi ? 8 : 0;
            #pragma unroll
            for (int r = 0; r < 8; ++r) {
                _Float16* pw = ph + (rowb + r) * 72 + lm;
                pw[0]  = (_Float16)sA[r];
                pw[16] = (_Float16)sB[r];
                pw[32] = (_Float16)sC[r];
                pw[48] = (_Float16)sD[r];
            }
        }
        __syncthreads();
        V16U p0, p1;
        const _Float16* pr = ph + lm * 72;
        p0.h[0] = *(const v8h*)(pr + (hi ? 16 : 0));     // K (cols) 0-7  | 16-23
        p0.h[1] = *(const v8h*)(pr + (hi ? 24 : 8));     // K 8-15 | 24-31
        p1.h[0] = *(const v8h*)(pr + (hi ? 48 : 32));    // K 32-39 | 48-55
        p1.h[1] = *(const v8h*)(pr + (hi ? 56 : 40));    // K 40-47 | 56-63

        // ---- PV + row-sum accumulation ----
        v16h bv00 = *(const v16h*)(vb + (size_t)(lm)      * SS + t0      + (hi ? 16 : 0));
        v16h bv01 = *(const v16h*)(vb + (size_t)(lm)      * SS + t0 + 32 + (hi ? 16 : 0));
        v16h bv10 = *(const v16h*)(vb + (size_t)(16 + lm) * SS + t0      + (hi ? 16 : 0));
        v16h bv11 = *(const v16h*)(vb + (size_t)(16 + lm) * SS + t0 + 32 + (hi ? 16 : 0));
        acc0 = __builtin_amdgcn_wmma_f32_16x16x32_f16(false, p0.v, false, bv00, (short)0, acc0, false, false);
        acc0 = __builtin_amdgcn_wmma_f32_16x16x32_f16(false, p1.v, false, bv01, (short)0, acc0, false, false);
        acc1 = __builtin_amdgcn_wmma_f32_16x16x32_f16(false, p0.v, false, bv10, (short)0, acc1, false, false);
        acc1 = __builtin_amdgcn_wmma_f32_16x16x32_f16(false, p1.v, false, bv11, (short)0, acc1, false, false);
        accL = __builtin_amdgcn_wmma_f32_16x16x32_f16(false, p0.v, false, ones, (short)0, accL, false, false);
        accL = __builtin_amdgcn_wmma_f32_16x16x32_f16(false, p1.v, false, ones, (short)0, accL, false, false);
    }

    // Normalize and write: D element r -> row s0+r+8*hi, column c = lane
    float* zn = z + (size_t)n * DM * SS;
    #pragma unroll
    for (int r = 0; r < 8; ++r) {
        const int srow = s0 + r + (hi ? 8 : 0);
        const float inv = 1.0f / accL[r];        // row sum replicated to all cols
        zn[(size_t)(lm)      * SS + srow] = acc0[r] * inv;
        zn[(size_t)(16 + lm) * SS + srow] = acc1[r] * inv;
    }
}

// ---------------------------------------------------------------------------
// Kernel 3: Z = Wz @ concat(Zh,Zm) + bz ; x = concat(Z, h)   (x: [n][96][s] f32)
// ---------------------------------------------------------------------------
__global__ __launch_bounds__(256) void zmix_kernel(
    const float* __restrict__ Zh, const float* __restrict__ Zm,
    const float* __restrict__ h,  const float* __restrict__ Wz,
    const float* __restrict__ bz, float* __restrict__ xb)
{
    const int idx = blockIdx.x * 256 + threadIdx.x;   // (n,s)
    const int n   = idx >> 12;
    const int sp  = idx & (SS - 1);

    float zc[2 * DM];
    const float* zh = Zh + (size_t)n * DM * SS;
    const float* zm = Zm + (size_t)n * DM * SS;
    #pragma unroll
    for (int i = 0; i < DM; ++i) zc[i]      = zh[(size_t)i * SS + sp];
    #pragma unroll
    for (int i = 0; i < DM; ++i) zc[DM + i] = zm[(size_t)i * SS + sp];

    float* xo = xb + (size_t)n * (DM + CIN) * SS;
    for (int o = 0; o < DM; ++o) {
        float acc = bz[o];
        const float* wr = Wz + o * 2 * DM;
        #pragma unroll
        for (int i = 0; i < 2 * DM; ++i) acc += wr[i] * zc[i];
        xo[(size_t)o * SS + sp] = acc;
    }
    for (int c = 0; c < CIN; ++c)
        xo[(size_t)(DM + c) * SS + sp] = h[((size_t)(n * CIN + c)) * SS + sp];
}

// ---------------------------------------------------------------------------
// Kernel 4: 3x3 SAME conv (Cin=96 -> Co=192) fused with LSTM-style gates.
// Each thread: one pixel, output channels {c, c+64, c+128} = (i, g, o).
// ---------------------------------------------------------------------------
__global__ __launch_bounds__(256) void conv_gate_kernel(
    const float* __restrict__ xb, const float* __restrict__ m,
    const float* __restrict__ Wo, const float* __restrict__ bo,
    float* __restrict__ out_h, float* __restrict__ out_m)
{
    __shared__ float ws[3 * 96 * 9];           // three filter rows, 10368 B
    const int tid = threadIdx.x;
    const int b   = blockIdx.x;                // n*64*16 + c*16 + pb
    const int pb  = b & 15;
    const int c   = (b >> 4) & 63;
    const int n   = b >> 10;

    for (int i2 = tid; i2 < 3 * 96 * 9; i2 += 256) {
        const int j   = i2 / (96 * 9);         // which of {i,g,o}
        const int rem = i2 - j * (96 * 9);     // ci*9 + ky*3 + kx
        ws[i2] = Wo[((size_t)(c + j * 64)) * (96 * 9) + rem];
    }
    __syncthreads();

    const int p  = pb * 256 + tid;
    const int y  = p >> 6;
    const int xc = p & 63;

    float a0 = bo[c], a1 = bo[c + 64], a2 = bo[c + 128];
    const float* xn = xb + (size_t)n * 96 * SS;
    for (int ci = 0; ci < 96; ++ci) {
        const float* xr = xn + (size_t)ci * SS;
        const float* wj = &ws[ci * 9];
        #pragma unroll
        for (int ky = 0; ky < 3; ++ky) {
            const int yy = y + ky - 1;
            if (yy < 0 || yy >= HW) continue;
            #pragma unroll
            for (int kx = 0; kx < 3; ++kx) {
                const int xx = xc + kx - 1;
                if (xx < 0 || xx >= HW) continue;
                const float v = xr[yy * HW + xx];
                a0 += v * wj[ky * 3 + kx];
                a1 += v * wj[96 * 9 + ky * 3 + kx];
                a2 += v * wj[2 * 96 * 9 + ky * 3 + kx];
            }
        }
    }

    const size_t oidx = ((size_t)(n * CIN + c)) * SS + p;
    const float ig = 1.0f / (1.0f + __expf(-a0));   // sigmoid(i)
    const float gg = tanhf(a1);
    const float og = 1.0f / (1.0f + __expf(-a2));   // sigmoid(o)
    const float m_old = m[oidx];
    const float m_next = ig * gg + (1.0f - ig) * m_old;
    out_m[oidx] = m_next;
    out_h[oidx] = og * m_next;
}

// ---------------------------------------------------------------------------
extern "C" void kernel_launch(void* const* d_in, const int* in_sizes, int n_in,
                              void* d_out, int out_size, void* d_ws, size_t ws_size,
                              hipStream_t stream) {
    const float* h  = (const float*)d_in[0];
    const float* m  = (const float*)d_in[1];
    const float* Wh = (const float*)d_in[2];
    const float* bh = (const float*)d_in[3];
    const float* Wm = (const float*)d_in[4];
    const float* bm = (const float*)d_in[5];
    const float* Wz = (const float*)d_in[6];
    const float* bz = (const float*)d_in[7];
    const float* Wo = (const float*)d_in[8];
    const float* bo = (const float*)d_in[9];

    // Workspace layout (~15 MB total)
    _Float16* q  = (_Float16*)d_ws;                 // [4][4096][32]  1 MB
    _Float16* kh = q  + (size_t)NS * DM;            // [4][4096][32]  1 MB
    _Float16* km = kh + (size_t)NS * DM;            // [4][4096][32]  1 MB
    _Float16* vh = km + (size_t)NS * DM;            // [4][32][4096]  1 MB
    _Float16* vm = vh + (size_t)NS * DM;            // [4][32][4096]  1 MB
    float* Zh = (float*)(vm + (size_t)NS * DM);     // [4][32][4096]  2 MB
    float* Zm = Zh + (size_t)BN * DM * SS;          // [4][32][4096]  2 MB
    float* xb = Zm + (size_t)BN * DM * SS;          // [4][96][4096]  6 MB

    float* out_h = (float*)d_out;
    float* out_m = out_h + (size_t)BN * CIN * SS;

    proj_kernel<<<NS / 128, 128, 0, stream>>>(h, m, Wh, bh, Wm, bm, q, kh, vh, km, vm);
    attn_kernel<<<BN * (SS / 16), 32, 0, stream>>>(q, kh, vh, Zh);
    attn_kernel<<<BN * (SS / 16), 32, 0, stream>>>(q, km, vm, Zm);
    zmix_kernel<<<NS / 256, 256, 0, stream>>>(Zh, Zm, h, Wz, bz, xb);
    conv_gate_kernel<<<BN * 64 * (SS / 256), 256, 0, stream>>>(xb, m, Wo, bo, out_h, out_m);
}
